// MistralFlashAttention_KVmix_32478542692826
// MI455X (gfx1250) — matrix operand order repaired and checked by reading, not compiled, not run on hardware
//
#include <hip/hip_runtime.h>
#include <math.h>

// ---------------- types ----------------
typedef __bf16 bf16;
typedef __attribute__((ext_vector_type(16))) __bf16 v16bf;
typedef __attribute__((ext_vector_type(8)))  float  v8f;
typedef unsigned int u32;
typedef __attribute__((ext_vector_type(4))) u32   u32x4;
typedef __attribute__((ext_vector_type(4))) float f32x4;

// ---------------- problem constants ----------------
#define NHQ   32
#define NKVH  8
#define HDIM  128
#define QLEN  512
#define HID_  4096
#define PASTL 3584
#define KVLEN 4096
#define GSZ   32

union Frag { u32x4 q[2]; v16bf v; };

__device__ __forceinline__ bf16 tobf(float f) {
  union { float f; u32 u; } a; a.f = f;
  u32 u = a.u + 0x7FFFu + ((a.u >> 16) & 1u);
  union { unsigned short s; bf16 b; } o; o.s = (unsigned short)(u >> 16);
  return o.b;
}

__device__ __forceinline__ v8f wmma_bf16(v16bf a, v16bf b, v8f c) {
  return __builtin_amdgcn_wmma_f32_16x16x32_bf16(false, a, false, b, (short)0, c,
                                                 false, false);
}

// A-fragment (16x32 bf16): lane l holds row (l&15); K = base + (l&16?8:0) + {0..7,16..23}
__device__ __forceinline__ v16bf frag_a(const bf16* tile, int ld, int row0, int k0, int lane) {
  const bf16* r = tile + (size_t)(row0 + (lane & 15)) * ld + k0 + ((lane & 16) ? 8 : 0);
  Frag f;
  f.q[0] = *(const u32x4*)(r);
  f.q[1] = *(const u32x4*)(r + 16);
  return f.v;
}

// B-fragment (32x16 bf16): lane l holds column (l&15); K = base + (l&16?8:0) + {0..7,16..23}
// "tile" rows are the B-matrix columns (i.e. W rows / K rows / V^T rows), K contiguous.
__device__ __forceinline__ v16bf frag_b(const bf16* tile, int ld, int row0, int k0, int lane) {
  const bf16* r = tile + (size_t)(row0 + (lane & 15)) * ld + k0 + ((lane & 16) ? 8 : 0);
  Frag f;
  f.q[0] = *(const u32x4*)(r);
  f.q[1] = *(const u32x4*)(r + 16);
  return f.v;
}

__device__ __forceinline__ void pack8(bf16* dst, f32x4 lo, f32x4 hi) {
  union { bf16 b[8]; u32x4 u; } t;
  t.b[0] = tobf(lo.x); t.b[1] = tobf(lo.y); t.b[2] = tobf(lo.z); t.b[3] = tobf(lo.w);
  t.b[4] = tobf(hi.x); t.b[5] = tobf(hi.y); t.b[6] = tobf(hi.z); t.b[7] = tobf(hi.w);
  *(u32x4*)dst = t.u;
}

// gfx1250 async copy: global -> LDS, 16B per lane, tracked by ASYNCcnt
__device__ __forceinline__ void async_g2l_b128(const bf16* gaddr, bf16* lds) {
  u32 laddr = (u32)(size_t)lds;  // low 32 bits of generic shared ptr == LDS offset
  asm volatile("global_load_async_to_lds_b128 %0, %1, off"
               :: "v"(laddr), "v"(gaddr) : "memory");
}
__device__ __forceinline__ void wait_async(void) {
  asm volatile("s_wait_asynccnt 0x0" ::: "memory");
}
__device__ __forceinline__ void wait_async16(void) {
  asm volatile("s_wait_asynccnt 0x10" ::: "memory");  // allow 16 in flight (next tile)
}

// ---------------------------------------------------------------------------
// Generic GEMM: C(MxN) = A(MxK) * W(NxK)^T.  A,W fp32 row-major (K contiguous),
// converted to bf16 while staging into LDS; accumulate fp32 via WMMA.
// Block tile 64x128, 256 threads = 8 waves, each wave 32x32 (2x2 WMMA tiles).
// ---------------------------------------------------------------------------
__global__ __launch_bounds__(256) void gemm_wmma_bf16(
    const float* __restrict__ A, const float* __restrict__ W,
    float* __restrict__ C, int M, int N, int K) {
  __shared__ __attribute__((aligned(16))) bf16 As[64 * 32];
  __shared__ __attribute__((aligned(16))) bf16 Ws[128 * 32];

  const int tid  = threadIdx.x;
  const int lane = tid & 31;
  const int wid  = tid >> 5;
  const int wr   = wid & 1;   // 0..1 : 32-row half
  const int wc   = wid >> 1;  // 0..3 : 32-col quarter
  const int m0 = blockIdx.y * 64;
  const int n0 = blockIdx.x * 128;

  v8f acc[2][2];
  for (int i = 0; i < 2; i++)
    for (int j = 0; j < 2; j++)
      for (int e = 0; e < 8; e++) acc[i][j][e] = 0.0f;

  const int arow  = tid >> 2;        // 0..63
  const int apart = (tid & 3) * 8;   // 0,8,16,24
  const int wrow  = tid >> 1;        // 0..127
  const int whalf = (tid & 1) * 16;  // 0 or 16

  for (int kc = 0; kc < K; kc += 32) {
    const float* ap = A + (size_t)(m0 + arow) * K + kc + apart;
    const float* wp = W + (size_t)(n0 + wrow) * K + kc + whalf;
    if (kc + 32 < K) {                       // gfx1250 global_prefetch_b8
      __builtin_prefetch(ap + 32, 0, 1);
      __builtin_prefetch(wp + 32, 0, 1);
    }
    f32x4 a0 = *(const f32x4*)(ap);
    f32x4 a1 = *(const f32x4*)(ap + 4);
    f32x4 w0 = *(const f32x4*)(wp);
    f32x4 w1 = *(const f32x4*)(wp + 4);
    f32x4 w2 = *(const f32x4*)(wp + 8);
    f32x4 w3 = *(const f32x4*)(wp + 12);
    pack8(As + arow * 32 + apart,    a0, a1);
    pack8(Ws + wrow * 32 + whalf,     w0, w1);
    pack8(Ws + wrow * 32 + whalf + 8, w2, w3);
    __syncthreads();

    v16bf af[2], bfg[2];
    for (int mt = 0; mt < 2; mt++)
      af[mt] = frag_a(As, 32, wr * 32 + mt * 16, 0, lane);
    for (int nt = 0; nt < 2; nt++)
      bfg[nt] = frag_b(Ws, 32, wc * 32 + nt * 16, 0, lane);
    for (int mt = 0; mt < 2; mt++)
      for (int nt = 0; nt < 2; nt++)
        acc[mt][nt] = wmma_bf16(af[mt], bfg[nt], acc[mt][nt]);
    __syncthreads();
  }

  // C layout: VGPR r -> row r (lanes 0-15) / row r+8 (lanes 16-31), col = lane&15
  const int rbase = (lane & 16) ? 8 : 0;
  const int cn = lane & 15;
  for (int mt = 0; mt < 2; mt++)
    for (int nt = 0; nt < 2; nt++)
      for (int r = 0; r < 8; r++) {
        int rr = m0 + wr * 32 + mt * 16 + r + rbase;
        int cc = n0 + wc * 32 + nt * 16 + cn;
        C[(size_t)rr * N + cc] = acc[mt][nt][r];
      }
}

// ---------------------------------------------------------------------------
// 4-bit fake-quant of past K (groups of 32 along kv, per (h,d)) -> Kbuf (h,kv,d) bf16
// grid: (PASTL/32, NKVH), block: 128 (= one d per thread)
// ---------------------------------------------------------------------------
__global__ __launch_bounds__(128) void qdq_k_kernel(
    const float* __restrict__ pk, bf16* __restrict__ Kbuf) {
  const int g = blockIdx.x, h = blockIdx.y, d = threadIdx.x;
  const int kv0 = g * GSZ;
  const float* src = pk + ((size_t)h * PASTL + kv0) * HDIM + d;
  float v[GSZ];
  float mn = 1e30f, mx = -1e30f;
  for (int i = 0; i < GSZ; i++) {
    v[i] = src[(size_t)i * HDIM];
    mn = fminf(mn, v[i]);
    mx = fmaxf(mx, v[i]);
  }
  const float scale = (mx - mn) * (1.0f / 15.0f);
  const float inv = scale > 0.0f ? 1.0f / scale : 0.0f;
  bf16* dst = Kbuf + ((size_t)h * KVLEN + kv0) * HDIM + d;
  for (int i = 0; i < GSZ; i++) {
    float q = rintf((v[i] - mn) * inv);
    q = fminf(fmaxf(q, 0.0f), 15.0f);
    dst[(size_t)i * HDIM] = tobf(q * scale + mn);
  }
}

// ---------------------------------------------------------------------------
// 4-bit fake-quant of past V (groups of 32 along d, per (h,kv)) -> VT (h,d,kv) bf16
// one thread per group; grid*block == NKVH*PASTL*4 exactly
// ---------------------------------------------------------------------------
__global__ __launch_bounds__(256) void qdq_v_kernel(
    const float* __restrict__ pv, bf16* __restrict__ VT) {
  const int idx = blockIdx.x * 256 + threadIdx.x;
  const int h  = idx / (PASTL * 4);
  const int r  = idx % (PASTL * 4);
  const int kv = r >> 2;
  const int dg = (r & 3) * GSZ;
  const float* src = pv + ((size_t)h * PASTL + kv) * HDIM + dg;
  float v[GSZ];
  float mn = 1e30f, mx = -1e30f;
  for (int i = 0; i < GSZ; i++) {
    v[i] = src[i];
    mn = fminf(mn, v[i]);
    mx = fmaxf(mx, v[i]);
  }
  const float scale = (mx - mn) * (1.0f / 15.0f);
  const float inv = scale > 0.0f ? 1.0f / scale : 0.0f;
  bf16* dst = VT + ((size_t)h * HDIM + dg) * KVLEN + kv;
  for (int i = 0; i < GSZ; i++) {
    float q = rintf((v[i] - mn) * inv);
    q = fminf(fmaxf(q, 0.0f), 15.0f);
    dst[(size_t)i * KVLEN] = tobf(q * scale + mn);
  }
}

// ---------------------------------------------------------------------------
// RoPE Q & new K; scatter Q->(h,q,d) bf16, K->Kbuf[h][PAST+i][d], V->VT[h][d][PAST+i]
// grid: QLEN blocks (one query row each), 256 threads
// ---------------------------------------------------------------------------
__global__ __launch_bounds__(256) void rope_scatter_kernel(
    const float* __restrict__ Qf, const float* __restrict__ Kf,
    const float* __restrict__ Vf, bf16* __restrict__ Qb,
    bf16* __restrict__ Kbuf, bf16* __restrict__ VT) {
  const int i = blockIdx.x;
  const int tid = threadIdx.x;
  const float pos = (float)(PASTL + i);
  const float LTH = 0.14391156f;  // ln(10000)/64

  // Q: 32 heads * 64 rotation pairs
  for (int p = tid; p < NHQ * 64; p += 256) {
    const int hh = p >> 6, j = p & 63;
    const float f = __expf(-(float)j * LTH) * pos;
    float s, c;
    __sincosf(f, &s, &c);
    const float x1 = Qf[(size_t)i * HID_ + hh * HDIM + j];
    const float x2 = Qf[(size_t)i * HID_ + hh * HDIM + j + 64];
    bf16* qb = Qb + ((size_t)hh * QLEN + i) * HDIM;
    qb[j]      = tobf(x1 * c - x2 * s);
    qb[j + 64] = tobf(x2 * c + x1 * s);
  }
  // K: 8 heads * 64 pairs
  for (int p = tid; p < NKVH * 64; p += 256) {
    const int hh = p >> 6, j = p & 63;
    const float f = __expf(-(float)j * LTH) * pos;
    float s, c;
    __sincosf(f, &s, &c);
    const float x1 = Kf[(size_t)i * (NKVH * HDIM) + hh * HDIM + j];
    const float x2 = Kf[(size_t)i * (NKVH * HDIM) + hh * HDIM + j + 64];
    bf16* kb = Kbuf + ((size_t)hh * KVLEN + PASTL + i) * HDIM;
    kb[j]      = tobf(x1 * c - x2 * s);
    kb[j + 64] = tobf(x2 * c + x1 * s);
  }
  // V: 8 heads * 128 dims (no rope), transposed store
  for (int p = tid; p < NKVH * HDIM; p += 256) {
    const int hh = p >> 7, d = p & 127;
    VT[((size_t)hh * HDIM + d) * KVLEN + PASTL + i] =
        tobf(Vf[(size_t)i * (NKVH * HDIM) + p]);
  }
}

// ---------------------------------------------------------------------------
// Flash attention: block = (64-query tile, head), 128 threads = 4 waves x 16 rows.
// S = Q K^T (bf16 WMMA, f32 acc), online softmax, O += P V.
// K/V tiles double-buffered in LDS via GLOBAL_LOAD_ASYNC_TO_LDS_B128 (ASYNCcnt).
// Kbuf: (h, kv, d) bf16 ; VT: (h, d, kv) bf16 ; Qb: (h, q, d) bf16
// ---------------------------------------------------------------------------
__global__ __launch_bounds__(128) void attn_kernel(
    const bf16* __restrict__ Qb, const bf16* __restrict__ Kbuf,
    const bf16* __restrict__ VT, float* __restrict__ O) {
  const int qt  = blockIdx.x;   // 0..7
  const int h   = blockIdx.y;   // 0..31
  const int kvh = h >> 2;       // GQA: 4 q-heads per kv-head
  const int tid = threadIdx.x;
  const int lane = tid & 31, wv = tid >> 5;

  __shared__ __attribute__((aligned(16))) bf16 Ks[2][64 * 128];   // (kv, d)
  __shared__ __attribute__((aligned(16))) bf16 Vs[2][128 * 64];   // (d, kv)
  __shared__ __attribute__((aligned(16))) bf16 Ps[4][16 * 64];    // per-wave P

  const bf16* kbase = Kbuf + (size_t)kvh * KVLEN * HDIM;
  const bf16* vbase = VT + (size_t)kvh * HDIM * KVLEN;

  // issue one kv-tile's async copies (16 instructions per thread-block pass;
  // each thread issues 8+8? no: 8 K + 8 V spread over the block => 16 per wave)
  auto issue_tile = [&](int t, int buf) {
    const bf16* kg = kbase + (size_t)(t * 64) * HDIM;
    for (int i = 0; i < 8; i++) {
      int idx = i * 128 + tid;                  // 1024 x 16B chunks
      int r = idx >> 4, c8 = (idx & 15) * 8;
      async_g2l_b128(kg + (size_t)r * HDIM + c8, &Ks[buf][r * 128 + c8]);
    }
    const bf16* vg = vbase + t * 64;
    for (int i = 0; i < 8; i++) {
      int idx = i * 128 + tid;
      int r = idx >> 3, c8 = (idx & 7) * 8;
      async_g2l_b128(vg + (size_t)r * KVLEN + c8, &Vs[buf][r * 64 + c8]);
    }
  };

  // Q fragments for this wave's 16 rows (loaded once)
  const int qrow_l = qt * 64 + wv * 16 + (lane & 15);
  const bf16* qrp = Qb + ((size_t)h * QLEN + qrow_l) * HDIM;
  v16bf qf[4];
  for (int c = 0; c < 4; c++) {
    Frag f;
    const bf16* p = qrp + c * 32 + ((lane & 16) ? 8 : 0);
    f.q[0] = *(const u32x4*)(p);
    f.q[1] = *(const u32x4*)(p + 16);
    qf[c] = f.v;
  }

  float mrow[8], lrow[8];
  for (int r = 0; r < 8; r++) { mrow[r] = -1e30f; lrow[r] = 0.0f; }
  v8f o[8];
  for (int dt = 0; dt < 8; dt++)
    for (int e = 0; e < 8; e++) o[dt][e] = 0.0f;

  const int rbase = (lane & 16) ? 8 : 0;
  const int cn = lane & 15;
  const int qg = qt * 64 + wv * 16 + rbase;  // global q row for VGPR r is qg+r
  const float sc = 0.08838834764831845f;     // 1/sqrt(128)
  const int tEnd = PASTL / 64 + qt;          // last (partially masked) kv tile

  issue_tile(0, 0);

  for (int t = 0; t <= tEnd; ++t) {
    const int buf = t & 1;
    if (t < tEnd) {
      issue_tile(t + 1, buf ^ 1);  // prefetch next tile into other buffer
      wait_async16();              // tile t's 16 copies done; next 16 in flight
    } else {
      wait_async();
    }
    __syncthreads();

    const bf16* KsT = Ks[buf];
    const bf16* VsT = Vs[buf];

    // S = Q K^T  (16x64 per wave)
    v8f s[4];
    for (int nt = 0; nt < 4; nt++)
      for (int e = 0; e < 8; e++) s[nt][e] = 0.0f;
    for (int nt = 0; nt < 4; nt++)
      for (int c = 0; c < 4; c++) {
        v16bf b = frag_b(KsT, 128, nt * 16, c * 32, lane);
        s[nt] = wmma_bf16(qf[c], b, s[nt]);
      }

    // scale + causal mask
    for (int nt = 0; nt < 4; nt++) {
      const int col = t * 64 + nt * 16 + cn;
      for (int r = 0; r < 8; r++) {
        float sv = s[nt][r] * sc;
        if (col > PASTL + qg + r) sv = -1e30f;
        s[nt][r] = sv;
      }
    }

    // online softmax (row reductions across the 16-lane half-wave)
    float alpha[8];
    for (int r = 0; r < 8; r++) {
      float cm = fmaxf(fmaxf(s[0][r], s[1][r]), fmaxf(s[2][r], s[3][r]));
      cm = fmaxf(cm, __shfl_xor(cm, 1));
      cm = fmaxf(cm, __shfl_xor(cm, 2));
      cm = fmaxf(cm, __shfl_xor(cm, 4));
      cm = fmaxf(cm, __shfl_xor(cm, 8));
      const float mnew = fmaxf(mrow[r], cm);
      alpha[r] = __expf(mrow[r] - mnew);
      mrow[r] = mnew;
      float rs = 0.0f;
      bf16* prow = Ps[wv] + (r + rbase) * 64;
      for (int nt = 0; nt < 4; nt++) {
        const float p = __expf(s[nt][r] - mnew);
        prow[nt * 16 + cn] = tobf(p);
        rs += p;
      }
      rs += __shfl_xor(rs, 1);
      rs += __shfl_xor(rs, 2);
      rs += __shfl_xor(rs, 4);
      rs += __shfl_xor(rs, 8);
      lrow[r] = lrow[r] * alpha[r] + rs;
      for (int dt = 0; dt < 8; dt++) o[dt][r] *= alpha[r];
    }
    // same-wave LDS RAW (Ps) is in-order; no barrier needed before re-fragmenting

    // O += P V   (P: 16x64 A-frags; V^T rows give kv-contiguous B-frags)
    for (int c = 0; c < 2; c++) {
      v16bf a = frag_a(Ps[wv], 64, 0, c * 32, lane);
      for (int dt = 0; dt < 8; dt++) {
        v16bf b = frag_b(VsT, 64, dt * 16, c * 32, lane);
        o[dt] = wmma_bf16(a, b, o[dt]);
      }
    }
    __syncthreads();  // all waves done reading buf before it is re-issued at t+1
  }

  // epilogue: normalize and write (q, h*128+d) fp32
  for (int r = 0; r < 8; r++) {
    const float invl = 1.0f / lrow[r];
    const int row = qg + r;
    float* orow = O + (size_t)row * HID_ + h * HDIM;
    for (int dt = 0; dt < 8; dt++)
      orow[dt * 16 + cn] = o[dt][r] * invl;
  }
}

// ---------------------------------------------------------------------------
extern "C" void kernel_launch(void* const* d_in, const int* in_sizes, int n_in,
                              void* d_out, int out_size, void* d_ws, size_t ws_size,
                              hipStream_t stream) {
  (void)in_sizes; (void)n_in; (void)out_size; (void)ws_size;
  const float* hidden     = (const float*)d_in[0];
  const float* past_key   = (const float*)d_in[1];
  const float* past_value = (const float*)d_in[2];
  const float* wq = (const float*)d_in[3];
  const float* wk = (const float*)d_in[4];
  const float* wv = (const float*)d_in[5];
  const float* wo = (const float*)d_in[6];
  // d_in[7] = position_ids (implicitly PAST..PAST+QL-1)

  char* ws = (char*)d_ws;
  size_t off = 0;
  float* Qf = (float*)(ws + off); off += (size_t)QLEN * HID_ * 4;          // 8 MB
  float* Kf = (float*)(ws + off); off += (size_t)QLEN * NKVH * HDIM * 4;   // 2 MB
  float* Vf = (float*)(ws + off); off += (size_t)QLEN * NKVH * HDIM * 4;   // 2 MB
  float* Of = (float*)(ws + off); off += (size_t)QLEN * HID_ * 4;          // 8 MB
  bf16* Qb   = (bf16*)(ws + off); off += (size_t)NHQ * QLEN * HDIM * 2;    // 4 MB
  bf16* Kbuf = (bf16*)(ws + off); off += (size_t)NKVH * KVLEN * HDIM * 2;  // 8 MB
  bf16* VT   = (bf16*)(ws + off); off += (size_t)NKVH * HDIM * KVLEN * 2;  // 8 MB

  // QKV projections (fp32 weights read once, bf16 WMMA)
  gemm_wmma_bf16<<<dim3(HID_ / 128, QLEN / 64), 256, 0, stream>>>(
      hidden, wq, Qf, QLEN, HID_, HID_);
  gemm_wmma_bf16<<<dim3((NKVH * HDIM) / 128, QLEN / 64), 256, 0, stream>>>(
      hidden, wk, Kf, QLEN, NKVH * HDIM, HID_);
  gemm_wmma_bf16<<<dim3((NKVH * HDIM) / 128, QLEN / 64), 256, 0, stream>>>(
      hidden, wv, Vf, QLEN, NKVH * HDIM, HID_);

  // fake-quant past cache into attention-friendly bf16 layouts
  qdq_k_kernel<<<dim3(PASTL / GSZ, NKVH), 128, 0, stream>>>(past_key, Kbuf);
  qdq_v_kernel<<<(NKVH * PASTL * 4) / 256, 256, 0, stream>>>(past_value, VT);

  // RoPE + append new K/V, scatter Q
  rope_scatter_kernel<<<QLEN, 256, 0, stream>>>(Qf, Kf, Vf, Qb, Kbuf, VT);

  // flash attention (async double-buffered K/V staging)
  attn_kernel<<<dim3(QLEN / 64, NHQ), 128, 0, stream>>>(Qb, Kbuf, VT, Of);

  // output projection
  gemm_wmma_bf16<<<dim3(HID_ / 128, QLEN / 64), 256, 0, stream>>>(
      Of, wo, (float*)d_out, QLEN, HID_, HID_);
}
